// TDNN_86388972191900
// MI455X (gfx1250) — matrix operand order repaired and checked by reading, not compiled, not run on hardware
//
#include <hip/hip_runtime.h>
#include <cstdint>
#include <cstddef>

// ---------------------------------------------------------------------------
// Types for CDNA5 WMMA
// ---------------------------------------------------------------------------
typedef __attribute__((ext_vector_type(16))) __bf16 v16bf;
typedef __attribute__((ext_vector_type(8)))  float  v8f;

#define BN_EPS 1e-5f
#define BATCH 32

__device__ __forceinline__ unsigned short f2bf(float f) {
  union { float f; unsigned u; } x; x.f = f;
  unsigned u = x.u;
  return (unsigned short)((u + 0x7FFFu + ((u >> 16) & 1u)) >> 16);  // RNE
}
__device__ __forceinline__ float bf2f(unsigned short h) {
  union { unsigned u; float f; } x; x.u = ((unsigned)h) << 16; return x.f;
}

// ---------------------------------------------------------------------------
// Prep kernels
// ---------------------------------------------------------------------------
// x: [B, T0, F0] f32  ->  xT: [B, FPAD, T0] bf16 (rows F0..FPAD-1 pre-zeroed)
__global__ void transpose_x_kernel(const float* __restrict__ x,
                                   unsigned short* __restrict__ xT,
                                   int T0, int F0, int FPAD) {
  int idx = blockIdx.x * blockDim.x + threadIdx.x;
  int total = BATCH * F0 * T0;
  if (idx >= total) return;
  int t = idx % T0;
  int f = (idx / T0) % F0;
  int b = idx / (T0 * F0);
  xT[((size_t)b * FPAD + f) * T0 + t] = f2bf(x[((size_t)b * T0 + t) * F0 + f]);
}

// Weight permute+pad:  src f32 [rows, F*C] with column k = f*C + c
//                  ->  dst bf16 [Mpad, C*Kc] with column k' = c*Kc + f   (zero pad)
__global__ void convert_weights_kernel(const float* __restrict__ src,
                                       unsigned short* __restrict__ dst,
                                       int rows, int F, int C, int Kc, int Mpad) {
  int Kpad = C * Kc;
  int idx = blockIdx.x * blockDim.x + threadIdx.x;
  int total = Mpad * Kpad;
  if (idx >= total) return;
  int m  = idx / Kpad;
  int kp = idx - m * Kpad;
  int c  = kp / Kc;
  int f  = kp - c * Kc;
  float v = (m < rows && f < F) ? src[(size_t)m * (F * C) + f * C + c] : 0.0f;
  dst[idx] = f2bf(v);
}

// ---------------------------------------------------------------------------
// TDNN stage GEMM (shape-templated so all loop loads use immediate offsets).
// out[b,m,t] = act( sum_{c,f} W'[m, c*KC+f] * in[b, f, t + c*STEP] + bias[m] )
// Block = 128 threads (4 waves). Block tile = 128M x 64N; each wave owns a
// 128M x 16N strip: one B fragment shared by 8 WMMAs (A = contiguous weights).
// Epilogue: bias + activation, optional bf16 store, per-(b,m) sum/sumsq stats.
// ---------------------------------------------------------------------------
template <int TIN, int TN, int C, int STEP, int KC>
__global__ __launch_bounds__(128)
void tdnn_wmma_gemm(const unsigned short* __restrict__ in,   // [B, Frows, TIN] bf16
                    const unsigned short* __restrict__ W,    // [Mpad, C*KC] bf16 (permuted)
                    const float* __restrict__ bias,          // [Cout]
                    unsigned short* __restrict__ out,        // [B, Cout, TN] bf16 or null
                    float* __restrict__ S1,                  // [B*Cout]
                    float* __restrict__ S2,                  // [B*Cout]
                    int Frows, int Cout, float actMax, int storeOut) {
  constexpr int KPAD = C * KC;
  const int b    = blockIdx.z;
  const int m0   = blockIdx.y << 7;           // 128 M rows per block
  const int wave = threadIdx.x >> 5;
  const int lane = threadIdx.x & 31;
  const int hh   = lane >> 4;                 // half-wave 0/1
  const int l16  = lane & 15;
  const int tg   = (blockIdx.x << 6) + (wave << 4) + l16;   // output time index
  const bool tvalid = (tg < TN);
  const int tc   = tvalid ? tg : (TN - 1);    // clamped load column (in bounds)

  const unsigned short* __restrict__ inB = in + (size_t)b * Frows * TIN;
  const int kb = hh << 3;                     // B-fragment K base per half-wave

  v8f acc[8] = {};

  const unsigned short* __restrict__ wrow =
      W + (size_t)(m0 + l16) * KPAD + (hh << 3);

#pragma unroll
  for (int c = 0; c < C; ++c) {
    const unsigned short* __restrict__ bp = inB + (size_t)kb * TIN + tc + c * STEP;
    const unsigned short* __restrict__ wp = wrow + c * KC;
    for (int f0 = 0; f0 < KC; f0 += 32) {
      // ---- B fragment: input rows (f0+kb)+0..7 and +16..23, fixed time column ----
      union { unsigned u[8]; v16bf v; } Bf;
#pragma unroll
      for (int v = 0; v < 8; ++v) {           // all offsets are compile-time imms
        unsigned lo = bp[(size_t)(2 * v + (v >= 4 ? 8 : 0)) * TIN];
        unsigned hi = bp[(size_t)(2 * v + 1 + (v >= 4 ? 8 : 0)) * TIN];
        Bf.u[v] = lo | (hi << 16);
      }
      __builtin_prefetch(bp + (size_t)32 * TIN, 0, 1);   // next K-chunk rows
      // ---- 8 M sub-tiles reuse this B fragment ----
#pragma unroll
      for (int s = 0; s < 8; ++s) {
        const unsigned short* ws = wp + (size_t)s * 16 * KPAD;  // imm offsets
        union { uint4 q[2]; v16bf v; } Af;
        Af.q[0] = *(const uint4*)(ws);
        Af.q[1] = *(const uint4*)(ws + 16);
        acc[s] = __builtin_amdgcn_wmma_f32_16x16x32_bf16(false, Af.v, false, Bf.v,
                                                         (short)0, acc[s], false, false);
      }
      bp += (size_t)32 * TIN;
      wp += 32;
    }
  }

  // ---- Epilogue ----
#pragma unroll
  for (int s = 0; s < 8; ++s) {
#pragma unroll
    for (int r = 0; r < 8; ++r) {
      int m = m0 + (s << 4) + r + (hh << 3);  // C/D layout: VGPR r -> M=r (+8 hi half)
      float y = 0.0f;
      if (tvalid && m < Cout) {
        y = acc[s][r] + bias[m];
        y = fminf(fmaxf(y, 0.0f), actMax);
        if (storeOut) out[((size_t)b * Cout + m) * TN + tg] = f2bf(y);
      }
      float p1 = y, p2 = y * y;
#pragma unroll
      for (int d = 1; d < 16; d <<= 1) {      // reduce across 16 lanes sharing m
        p1 += __shfl_xor(p1, d, 32);
        p2 += __shfl_xor(p2, d, 32);
      }
      if (l16 == 0 && m < Cout) {
        atomicAdd(&S1[b * Cout + m], p1);
        atomicAdd(&S2[b * Cout + m], p2);
      }
    }
  }
}

// ---------------------------------------------------------------------------
// BN stats finalize: scale[c] = g*rsqrt(var+eps), shift[c] = beta - mean*scale
// ---------------------------------------------------------------------------
__global__ void bn_stats_kernel(const float* __restrict__ S1,
                                const float* __restrict__ S2,
                                const float* __restrict__ gamma,
                                const float* __restrict__ beta,
                                float* __restrict__ scale,
                                float* __restrict__ shift,
                                int Cout, int Tn) {
  int c = blockIdx.x * blockDim.x + threadIdx.x;
  if (c >= Cout) return;
  float s1 = 0.f, s2 = 0.f;
  for (int b = 0; b < BATCH; ++b) { s1 += S1[b * Cout + c]; s2 += S2[b * Cout + c]; }
  float n = (float)BATCH * (float)Tn;
  float m = s1 / n;
  float var = fmaxf(s2 / n - m * m, 0.0f);
  float rs = rsqrtf(var + BN_EPS);
  float g  = gamma ? gamma[c] : 1.0f;
  float bb = beta ? beta[c] : 0.0f;
  scale[c] = rs * g;
  shift[c] = bb - m * rs * g;
}

// In-place BN application on a bf16 activation buffer [B, Cout, Tn]
__global__ void bn_apply_kernel(unsigned short* __restrict__ act,
                                const float* __restrict__ scale,
                                const float* __restrict__ shift,
                                int Cout, int Tn) {
  size_t idx = (size_t)blockIdx.x * blockDim.x + threadIdx.x;
  size_t total = (size_t)BATCH * Cout * Tn;
  if (idx >= total) return;
  int c = (int)((idx / Tn) % Cout);
  act[idx] = f2bf(bf2f(act[idx]) * scale[c] + shift[c]);
}

// ---------------------------------------------------------------------------
// Stats pooling of BN(h5): mean' = (S1/T)*scale+shift ; std' = std_unbiased*|scale|
// ---------------------------------------------------------------------------
__global__ void pool_stats_kernel(const float* __restrict__ S1,
                                  const float* __restrict__ S2,
                                  const float* __restrict__ scale,
                                  const float* __restrict__ shift,
                                  float* __restrict__ pooled,  // [B, 2*Cc]
                                  int Cc, int T) {
  int idx = blockIdx.x * blockDim.x + threadIdx.x;
  if (idx >= BATCH * Cc) return;
  int b = idx / Cc, c = idx - b * Cc;
  float s1 = S1[b * Cc + c], s2 = S2[b * Cc + c];
  float mean = s1 / (float)T;
  pooled[(size_t)b * 2 * Cc + c] = mean * scale[c] + shift[c];
  float varu = fmaxf((s2 - s1 * s1 / (float)T) / (float)(T - 1), 0.0f);
  pooled[(size_t)b * 2 * Cc + Cc + c] = sqrtf(varu) * fabsf(scale[c]);
}

// ---------------------------------------------------------------------------
// Small FC + relu6 :  out[b,o] = relu6( in[b,:] . w[o,:] + bias[o] )
// ---------------------------------------------------------------------------
__global__ void fc_relu6_kernel(const float* __restrict__ in,
                                const float* __restrict__ w,
                                const float* __restrict__ bias,
                                float* __restrict__ out, int K, int O) {
  int idx = blockIdx.x * blockDim.x + threadIdx.x;
  if (idx >= BATCH * O) return;
  int b = idx & (BATCH - 1);
  int o = idx >> 5;                 // BATCH == 32
  const float* wr = w + (size_t)o * K;
  const float* xr = in + (size_t)b * K;
  float acc = bias[o];
  for (int k = 0; k < K; ++k) acc += wr[k] * xr[k];
  out[(size_t)b * O + o] = fminf(fmaxf(acc, 0.0f), 6.0f);
}

// BatchNorm over the batch dim of [B, O]
__global__ void bn_batch_kernel(const float* __restrict__ in,
                                const float* __restrict__ gamma,
                                const float* __restrict__ beta,
                                float* __restrict__ out, int O) {
  int o = blockIdx.x * blockDim.x + threadIdx.x;
  if (o >= O) return;
  float s1 = 0.f, s2 = 0.f;
  for (int b = 0; b < BATCH; ++b) { float v = in[(size_t)b * O + o]; s1 += v; s2 += v * v; }
  float m = s1 / (float)BATCH;
  float var = fmaxf(s2 / (float)BATCH - m * m, 0.0f);
  float rs = rsqrtf(var + BN_EPS);
  for (int b = 0; b < BATCH; ++b)
    out[(size_t)b * O + o] = (in[(size_t)b * O + o] - m) * rs * gamma[o] + beta[o];
}

// ---------------------------------------------------------------------------
// Host driver
// ---------------------------------------------------------------------------
extern "C" void kernel_launch(void* const* d_in, const int* in_sizes, int n_in,
                              void* d_out, int out_size, void* d_ws, size_t ws_size,
                              hipStream_t stream) {
  (void)in_sizes; (void)n_in; (void)out_size; (void)ws_size;
  const float* x    = (const float*)d_in[0];
  const float* h1w  = (const float*)d_in[1];  const float* h1b = (const float*)d_in[2];
  const float* h2w  = (const float*)d_in[3];  const float* h2b = (const float*)d_in[4];
  const float* bn2g = (const float*)d_in[5];  const float* bn2b = (const float*)d_in[6];
  const float* h3w  = (const float*)d_in[7];  const float* h3b = (const float*)d_in[8];
  const float* bn3g = (const float*)d_in[9];  const float* bn3b = (const float*)d_in[10];
  const float* h4w  = (const float*)d_in[11]; const float* h4b = (const float*)d_in[12];
  const float* bn4g = (const float*)d_in[13]; const float* bn4b = (const float*)d_in[14];
  const float* h5w  = (const float*)d_in[15]; const float* h5b = (const float*)d_in[16];
  const float* bn5g = (const float*)d_in[17]; const float* bn5b = (const float*)d_in[18];
  const float* l1w  = (const float*)d_in[19]; const float* l1b = (const float*)d_in[20];
  const float* bn6g = (const float*)d_in[21]; const float* bn6b = (const float*)d_in[22];
  const float* l2w  = (const float*)d_in[23]; const float* l2b = (const float*)d_in[24];
  const float* bn7g = (const float*)d_in[25]; const float* bn7b = (const float*)d_in[26];

  char* ws = (char*)d_ws;
  size_t off = 0;
  auto alloc = [&](size_t bytes) -> void* {
    size_t a = (off + 255) & ~(size_t)255;
    off = a + bytes;
    return (void*)(ws + a);
  };

  unsigned short* xT   = (unsigned short*)alloc((size_t)32 * 32 * 2048 * 2);   // F padded 20->32
  unsigned short* w1p  = (unsigned short*)alloc((size_t)512 * 160 * 2);        // 5 * 32
  unsigned short* w2p  = (unsigned short*)alloc((size_t)512 * 1536 * 2);
  unsigned short* w3p  = (unsigned short*)alloc((size_t)512 * 1536 * 2);
  unsigned short* w4p  = (unsigned short*)alloc((size_t)512 * 512 * 2);
  unsigned short* w5p  = (unsigned short*)alloc((size_t)1536 * 512 * 2);       // M padded 1500->1536
  unsigned short* actA = (unsigned short*)alloc((size_t)32 * 512 * 2048 * 2);
  unsigned short* actB = (unsigned short*)alloc((size_t)32 * 512 * 2048 * 2);
  float* S1     = (float*)alloc((size_t)32 * 1536 * 4);
  float* S2     = (float*)alloc((size_t)32 * 1536 * 4);
  float* scaleB = (float*)alloc((size_t)1536 * 4);
  float* shiftB = (float*)alloc((size_t)1536 * 4);
  float* pooled = (float*)alloc((size_t)32 * 3000 * 4);
  float* t1     = (float*)alloc((size_t)32 * 512 * 4);
  float* t2     = (float*)alloc((size_t)32 * 512 * 4);
  float* t3     = (float*)alloc((size_t)32 * 512 * 4);

  // ---- prep: zero+transpose input, permute/convert weights to bf16 ----
  hipMemsetAsync(xT, 0, (size_t)32 * 32 * 2048 * 2, stream);
  {
    int tot = 32 * 20 * 2048;
    transpose_x_kernel<<<(tot + 255) / 256, 256, 0, stream>>>(x, xT, 2048, 20, 32);
  }
  auto convw = [&](const float* src, unsigned short* dst, int rows, int F, int C,
                   int Kc, int Mpad) {
    int tot = Mpad * C * Kc;
    convert_weights_kernel<<<(tot + 255) / 256, 256, 0, stream>>>(src, dst, rows, F, C, Kc, Mpad);
  };
  convw(h1w, w1p, 512, 20, 5, 32, 512);
  convw(h2w, w2p, 512, 512, 3, 512, 512);
  convw(h3w, w3p, 512, 512, 3, 512, 512);
  convw(h4w, w4p, 512, 512, 1, 512, 512);
  convw(h5w, w5p, 1500, 512, 1, 512, 1536);

  auto clear_stats = [&]() {
    hipMemsetAsync(S1, 0, (size_t)32 * 1536 * 4, stream);
    hipMemsetAsync(S2, 0, (size_t)32 * 1536 * 4, stream);
  };
  auto bn_apply = [&](unsigned short* act, int Cout, int Tn) {
    size_t tot = (size_t)32 * Cout * Tn;
    bn_apply_kernel<<<(unsigned)((tot + 255) / 256), 256, 0, stream>>>(act, scaleB, shiftB, Cout, Tn);
  };

  const float INF = 3.0e38f;

  // stage 1: splice(0..4) + h1 + relu ; BN (no affine)
  clear_stats();
  {
    dim3 g((2044 + 63) / 64, 512 / 128, 32);
    tdnn_wmma_gemm<2048, 2044, 5, 1, 32><<<g, 128, 0, stream>>>(
        xT, w1p, h1b, actA, S1, S2, 32, 512, INF, 1);
  }
  bn_stats_kernel<<<8, 64, 0, stream>>>(S1, S2, nullptr, nullptr, scaleB, shiftB, 512, 2044);
  bn_apply(actA, 512, 2044);

  // stage 2: splice(0,2,4) + h2 + relu ; BN(bn2)
  clear_stats();
  {
    dim3 g((2040 + 63) / 64, 512 / 128, 32);
    tdnn_wmma_gemm<2044, 2040, 3, 2, 512><<<g, 128, 0, stream>>>(
        actA, w2p, h2b, actB, S1, S2, 512, 512, INF, 1);
  }
  bn_stats_kernel<<<8, 64, 0, stream>>>(S1, S2, bn2g, bn2b, scaleB, shiftB, 512, 2040);
  bn_apply(actB, 512, 2040);

  // stage 3: splice(0,3,6) + h3 + relu ; BN(bn3)
  clear_stats();
  {
    dim3 g((2034 + 63) / 64, 512 / 128, 32);
    tdnn_wmma_gemm<2040, 2034, 3, 3, 512><<<g, 128, 0, stream>>>(
        actB, w3p, h3b, actA, S1, S2, 512, 512, INF, 1);
  }
  bn_stats_kernel<<<8, 64, 0, stream>>>(S1, S2, bn3g, bn3b, scaleB, shiftB, 512, 2034);
  bn_apply(actA, 512, 2034);

  // stage 4: h4 + relu6 ; BN(bn4)
  clear_stats();
  {
    dim3 g((2034 + 63) / 64, 512 / 128, 32);
    tdnn_wmma_gemm<2034, 2034, 1, 0, 512><<<g, 128, 0, stream>>>(
        actA, w4p, h4b, actB, S1, S2, 512, 512, 6.0f, 1);
  }
  bn_stats_kernel<<<8, 64, 0, stream>>>(S1, S2, bn4g, bn4b, scaleB, shiftB, 512, 2034);
  bn_apply(actB, 512, 2034);

  // stage 5: h5 + relu6 ; only per-(b,c) sums needed (BN + stats pooling fused)
  clear_stats();
  {
    dim3 g((2034 + 63) / 64, 1536 / 128, 32);
    tdnn_wmma_gemm<2034, 2034, 1, 0, 512><<<g, 128, 0, stream>>>(
        actB, w5p, h5b, nullptr, S1, S2, 512, 1500, 6.0f, 0);
  }
  bn_stats_kernel<<<(1500 + 63) / 64, 64, 0, stream>>>(S1, S2, bn5g, bn5b, scaleB, shiftB, 1500, 2034);
  {
    int tot = 32 * 1500;
    pool_stats_kernel<<<(tot + 255) / 256, 256, 0, stream>>>(S1, S2, scaleB, shiftB, pooled, 1500, 2034);
  }

  // FC head
  fc_relu6_kernel<<<(32 * 512 + 255) / 256, 256, 0, stream>>>(pooled, l1w, l1b, t1, 3000, 512);
  bn_batch_kernel<<<8, 64, 0, stream>>>(t1, bn6g, bn6b, t2, 512);
  fc_relu6_kernel<<<(32 * 512 + 255) / 256, 256, 0, stream>>>(t2, l2w, l2b, t3, 512, 512);
  bn_batch_kernel<<<8, 64, 0, stream>>>(t3, bn7g, bn7b, (float*)d_out, 512);
}
